// Attention_46119358825223
// MI455X (gfx1250) — matrix operand order repaired and checked by reading, not compiled, not run on hardware
//
#include <hip/hip_runtime.h>

// ---------- types ----------
typedef __attribute__((ext_vector_type(16))) __bf16       v16bf;
typedef __attribute__((ext_vector_type(8)))  float        v8f;
typedef __attribute__((ext_vector_type(4)))  float        f32x4;
typedef __attribute__((ext_vector_type(4)))  int          i32x4;
typedef __attribute__((ext_vector_type(4)))  unsigned int u32x4;
typedef __attribute__((ext_vector_type(2)))  unsigned int u32x2;

union BfVec { v16bf v; u32x4 u[2]; };

#define H 4
#define N 4096
#define D 256
#define NEGF (-9.0e15f)
#define LROW 8208u           // LDS row stride bytes: 4096*2 + 16B pad (bank skew, keeps 16B align)
#define ROWS 32              // output rows per workgroup (two 16-row A tiles share B loads)

__device__ __forceinline__ unsigned short f2bf(float f) {
  union { float f; unsigned u; } v; v.f = f;
  unsigned r = v.u + 0x7FFFu + ((v.u >> 16) & 1u);   // RNE
  return (unsigned short)(r >> 16);
}
__device__ __forceinline__ float bf2f(unsigned short u) {
  union { unsigned u; float f; } v; v.u = ((unsigned)u) << 16;
  return v.f;
}

// load one pipeline stage: two LDS A tiles + two global B tiles at K offset k0
__device__ __forceinline__ void ld_tiles(const unsigned short* eA0,
                                         const unsigned short* eA1,
                                         const unsigned short* xb0,
                                         const unsigned short* xb1,
                                         int k0, int half8,
                                         BfVec& A0, BfVec& A1, BfVec& B0, BfVec& B1) {
  const u32x4* ap0 = (const u32x4*)(eA0 + k0 + half8);   // 16B aligned
  A0.u[0] = ap0[0];                     // K = k0 + half*8 .. +7   -> vgpr 0..3
  A0.u[1] = ap0[2];                     // K = k0+16+half*8 .. +7  -> vgpr 4..7
  const u32x4* ap1 = (const u32x4*)(eA1 + k0 + half8);
  A1.u[0] = ap1[0];
  A1.u[1] = ap1[2];
  const u32x4* bp0 = (const u32x4*)(xb0 + k0);
  B0.u[0] = bp0[0]; B0.u[1] = bp0[2];   // K k0+half*8 .. +7, then k0+16+half*8 .. +7
  const u32x4* bp1 = (const u32x4*)(xb1 + k0);
  B1.u[0] = bp1[0]; B1.u[1] = bp1[2];
}

// 4 WMMAs; A0/A1 interleaved so both A tiles stay live in distinct registers
__device__ __forceinline__ void mm4(const BfVec& A0, const BfVec& A1,
                                    const BfVec& B0, const BfVec& B1,
                                    v8f& c00, v8f& c01, v8f& c10, v8f& c11) {
  c00 = __builtin_amdgcn_wmma_f32_16x16x32_bf16(false, A0.v, false, B0.v,
                                                (short)0, c00, false, false);
  c10 = __builtin_amdgcn_wmma_f32_16x16x32_bf16(false, A1.v, false, B0.v,
                                                (short)0, c10, false, false);
  c01 = __builtin_amdgcn_wmma_f32_16x16x32_bf16(false, A0.v, false, B1.v,
                                                (short)0, c01, false, false);
  c11 = __builtin_amdgcn_wmma_f32_16x16x32_bf16(false, A1.v, false, B1.v,
                                                (short)0, c11, false, false);
}

// ---------- kernel 1: x[H][N][D] fp32 -> xT[H][D][N] bf16 (tiled transpose) ----------
__global__ __launch_bounds__(256) void xpose_bf16(const float* __restrict__ x,
                                                  unsigned short* __restrict__ xt) {
  __shared__ unsigned short tile[32][33];
  const int h  = blockIdx.z;
  const int n0 = blockIdx.x * 32;
  const int d0 = blockIdx.y * 32;
  const int tx = threadIdx.x, ty = threadIdx.y;
  #pragma unroll
  for (int i = ty; i < 32; i += 8)
    tile[i][tx] = f2bf(x[(((size_t)h * N) + (n0 + i)) * D + d0 + tx]);
  __syncthreads();
  #pragma unroll
  for (int i = ty; i < 32; i += 8)
    xt[(((size_t)h * D) + (d0 + i)) * N + n0 + tx] = tile[tx][i];
}

// ---------- kernel 2: fused mask+leaky+softmax+GEMM, one (h, 32-row) block per WG ----------
__global__ __launch_bounds__(256) void gat_attn(const float* __restrict__ att,
                                                const int* __restrict__ adj,
                                                const unsigned short* __restrict__ xt,
                                                float* __restrict__ out) {
  extern __shared__ char smem[];
  // block order rb-major so the 4 heads sharing an adj strip run adjacently (adj L2 reuse)
  const int h    = blockIdx.x & 3;
  const int rb   = blockIdx.x >> 2;        // 0..127
  const int row0 = rb * ROWS;
  const int tid  = threadIdx.x;

  // ---- phase A/B mapping: 8 threads per row, 32 rows; each wave wholly owns 4 rows ----
  const int r = tid >> 3;                  // 0..31 row in block
  const int c = tid & 7;                   // 0..7 column-thread
  u32x2* eRow = (u32x2*)(smem + (size_t)r * LROW);        // 1024 u32x2 = 4096 bf16
  float* stats = (float*)(smem + (size_t)ROWS * LROW);    // [0..31] = 1/rowsum

  const f32x4* attRow = (const f32x4*)(att + (((size_t)h * N) + row0 + r) * N);
  const i32x4* adjRow = (const i32x4*)(adj + (size_t)(row0 + r) * N);

  // ---- Phase A: score = mask(leaky(a)); store bf16 to LDS; row max (over rounded) ----
  float m = NEGF;
  #pragma unroll 4
  for (int j = 0; j < 128; ++j) {
    const int q = j * 8 + c;               // coalesced: 8 threads x 16B = 128B/row
    f32x4 a = attRow[q];
    i32x4 g = adjRow[q];
    float s0 = (g.x > 0) ? ((a.x > 0.f) ? a.x : 0.2f * a.x) : NEGF;
    float s1 = (g.y > 0) ? ((a.y > 0.f) ? a.y : 0.2f * a.y) : NEGF;
    float s2 = (g.z > 0) ? ((a.z > 0.f) ? a.z : 0.2f * a.z) : NEGF;
    float s3 = (g.w > 0) ? ((a.w > 0.f) ? a.w : 0.2f * a.w) : NEGF;
    unsigned short b0 = f2bf(s0), b1 = f2bf(s1), b2 = f2bf(s2), b3 = f2bf(s3);
    m = fmaxf(m, fmaxf(fmaxf(bf2f(b0), bf2f(b1)), fmaxf(bf2f(b2), bf2f(b3))));
    u32x2 w; w.x = (unsigned)b0 | ((unsigned)b1 << 16);
             w.y = (unsigned)b2 | ((unsigned)b3 << 16);
    eRow[q] = w;
  }
  #pragma unroll
  for (int off = 4; off >= 1; off >>= 1)   // all 8 lanes of a row are in one wave
    m = fmaxf(m, __shfl_xor(m, off, 8));

  // ---- Phase B: e = exp(s - m) overwrite LDS; row sum (same thread's own slots) ----
  float sum = 0.f;
  #pragma unroll 4
  for (int j = 0; j < 128; ++j) {
    const int q = j * 8 + c;
    u32x2 w = eRow[q];
    float e0 = __expf(bf2f((unsigned short)(w.x & 0xFFFF)) - m);
    float e1 = __expf(bf2f((unsigned short)(w.x >> 16)) - m);
    float e2 = __expf(bf2f((unsigned short)(w.y & 0xFFFF)) - m);
    float e3 = __expf(bf2f((unsigned short)(w.y >> 16)) - m);
    sum += (e0 + e1) + (e2 + e3);
    u32x2 o; o.x = (unsigned)f2bf(e0) | ((unsigned)f2bf(e1) << 16);
             o.y = (unsigned)f2bf(e2) | ((unsigned)f2bf(e3) << 16);
    eRow[q] = o;
  }
  #pragma unroll
  for (int off = 4; off >= 1; off >>= 1) sum += __shfl_xor(sum, off, 8);
  if (c == 0) stats[r] = 1.0f / sum;
  __syncthreads();

  // ---- Phase C: out[32 x 256] = P[32 x 4096](bf16) @ X[4096 x 256](bf16), f32 acc ----
  // Two 16-row A tiles share every B tile -> 1 global b128 per WMMA.
  const int wv   = tid >> 5;               // 0..7 : wave owns d in [wv*32, wv*32+32)
  const int lane = tid & 31;
  const int rowA = lane & 15;
  const int half = lane >> 4;              // 0: K 0..7,16..23 / M 0..7 ; 1: K 8..15,24..31 / M 8..15
  const int half8 = half * 8;
  const int d0   = wv << 5;

  const unsigned short* eA0 = (const unsigned short*)(smem + (size_t)rowA * LROW);
  const unsigned short* eA1 = (const unsigned short*)(smem + (size_t)(rowA + 16) * LROW);
  const unsigned short* xb0 =
      xt + (((size_t)h * D) + d0 + (lane & 15)) * N + half * 8;
  const unsigned short* xb1 = xb0 + (size_t)16 * N;

  v8f acc00 = {}, acc01 = {}, acc10 = {}, acc11 = {};

  // explicit 2-deep software pipeline, two disjoint buffer sets (a/b):
  //   steady state: mm4(a); mm4(b); ld(a+64); ld(b+96)
  // mm4(b)'s 4 WMMAs give the >=4 co-exec slots needed between set-a's last WMMA
  // and the loads that overwrite set a -> no v_nop hazard stalls.
  BfVec aA0, aA1, aB0, aB1, bA0, bA1, bB0, bB1;
  ld_tiles(eA0, eA1, xb0, xb1, 0,  half8, aA0, aA1, aB0, aB1);
  ld_tiles(eA0, eA1, xb0, xb1, 32, half8, bA0, bA1, bB0, bB1);
  for (int k0 = 0; k0 <= N - 128; k0 += 64) {
    mm4(aA0, aA1, aB0, aB1, acc00, acc01, acc10, acc11);
    mm4(bA0, bA1, bB0, bB1, acc00, acc01, acc10, acc11);
    ld_tiles(eA0, eA1, xb0, xb1, k0 + 64, half8, aA0, aA1, aB0, aB1);
    ld_tiles(eA0, eA1, xb0, xb1, k0 + 96, half8, bA0, bA1, bB0, bB1);
  }
  mm4(aA0, aA1, aB0, aB1, acc00, acc01, acc10, acc11);
  mm4(bA0, bA1, bB0, bB1, acc00, acc01, acc10, acc11);

  // ---- epilogue: scale by 1/rowsum, store. C/D layout: vgpr g -> M = g + 8*half ----
  float* outRow = out + (((size_t)h * N) + row0) * D + d0 + (lane & 15);
  #pragma unroll
  for (int g = 0; g < 8; ++g) {
    const int rr   = g + half * 8;
    const float i0 = stats[rr];
    const float i1 = stats[16 + rr];
    outRow[(size_t)rr * D]             = acc00[g] * i0;
    outRow[(size_t)rr * D + 16]        = acc01[g] * i0;
    outRow[(size_t)(16 + rr) * D]      = acc10[g] * i1;
    outRow[(size_t)(16 + rr) * D + 16] = acc11[g] * i1;
  }
}

extern "C" void kernel_launch(void* const* d_in, const int* in_sizes, int n_in,
                              void* d_out, int out_size, void* d_ws, size_t ws_size,
                              hipStream_t stream) {
  const float* x   = (const float*)d_in[0];   // [H,N,D] f32
  const int*   adj = (const int*)d_in[1];     // [N,N] i32
  const float* att = (const float*)d_in[2];   // [H,N,N] f32
  float* out = (float*)d_out;                 // [H,N,D] f32
  unsigned short* xt = (unsigned short*)d_ws; // [H,D,N] bf16 (8 MB)

  xpose_bf16<<<dim3(N / 32, D / 32, H), dim3(32, 8), 0, stream>>>(x, xt);

  const size_t shmem = (size_t)ROWS * LROW + ROWS * sizeof(float); // ~257 KB of 320 KB WGP LDS
  (void)hipFuncSetAttribute((const void*)gat_attn,
                            hipFuncAttributeMaxDynamicSharedMemorySize, (int)shmem);
  gat_attn<<<dim3(H * (N / ROWS)), dim3(256), shmem, stream>>>(att, adj, xt, out);
}
